// GraphEncoder_57664230916996
// MI455X (gfx1250) — matrix operand order repaired and checked by reading, not compiled, not run on hardware
//
#include <hip/hip_runtime.h>

// ---------------- WMMA types / bf16 helpers ----------------
typedef __bf16 bf16_t;
typedef bf16_t v16bf __attribute__((ext_vector_type(16)));
typedef float  v8f   __attribute__((ext_vector_type(8)));

__device__ __forceinline__ bf16_t f2bf(float f) {
    unsigned u = __builtin_bit_cast(unsigned, f);
    unsigned r = u + 0x7fffu + ((u >> 16) & 1u);   // round-to-nearest-even
    unsigned short hs = (unsigned short)(r >> 16);
    return __builtin_bit_cast(bf16_t, hs);
}
__device__ __forceinline__ float bf2f(bf16_t h) {
    unsigned short hs = __builtin_bit_cast(unsigned short, h);
    unsigned u = ((unsigned)hs) << 16;
    return __builtin_bit_cast(float, u);
}

// ---------------- weight pre-swizzle into WMMA B-fragment order -----------
// For K-block kb (32 rows) and column tile nt (16 cols), fragment of 512 bf16:
//   elem[lane*16 + i] = W[kb*32 + (lane>>4)*16 + i][nt*16 + (lane&15)]
// so each lane's 16 values are one contiguous 32B v16bf load in the GEMM.
__global__ __launch_bounds__(256) void k_prep_w(
    const float* __restrict__ W, bf16_t* __restrict__ hi, bf16_t* __restrict__ lo,
    int K, int N)
{
    long t = (long)blockIdx.x * 256 + threadIdx.x;
    if (t >= (long)K * N) return;
    int  idx    = (int)(t & 511);
    long frag   = t >> 9;
    int  ntiles = N >> 4;
    int  nt     = (int)(frag % ntiles);
    int  kb     = (int)(frag / ntiles);
    int  l = idx >> 4, i = idx & 15;
    int  k = kb * 32 + (l >> 4) * 16 + i;
    int  n = nt * 16 + (l & 15);
    float v = W[(size_t)k * N + n];
    bf16_t h = f2bf(v);
    hi[t] = h;
    lo[t] = f2bf(v - bf2f(h));
}

// ---------------- GEMM: C[M,N] = A[M,K] @ B[K,N] (+bias, optional relu) ----
// 3xBF16 split precision, f32 accumulate. Each wave: 16 rows x 2 column tiles
// (A fragment converted once per K-step, reused for 6 v_wmma). B fragments
// come pre-swizzled in bf16 hi/lo planes (contiguous 32B per lane).
// Grid: x = N/256 (8 waves * 32 cols), y = ceil(M/16). K % 32 == 0.
__global__ __launch_bounds__(256) void k_gemm_bf16x3(
    const float* __restrict__ A,
    const bf16_t* __restrict__ Bhi, const bf16_t* __restrict__ Blo,
    const float* __restrict__ bias, float* __restrict__ C,
    int M, int K, int N, int fuse_relu)
{
    const int lane = threadIdx.x & 31;
    const int wave = threadIdx.x >> 5;
    const int m0 = blockIdx.y * 16;
    const int nt0 = blockIdx.x * 16 + wave * 2;   // first of 2 column tiles
    const int ntiles = N >> 4;

    // A fragment addressing (contiguous k = 16*(lane>>4) + i, same order as B)
    int am = m0 + (lane & 15);
    if (am >= M) am = M - 1;                       // M is a multiple of 16 here
    const int akb = (lane >> 4) * 16;

    v8f acc0 = {}, acc1 = {};
    for (int kk = 0; kk < K; kk += 32) {
        // ---- A: load 16 f32, split into bf16 hi/lo fragments ----
        const float4* ap0 = reinterpret_cast<const float4*>(A + (size_t)am * K + kk + akb);
        const float4* ap1 = reinterpret_cast<const float4*>(A + (size_t)am * K + kk + akb + 8);
        float4 a00 = ap0[0], a01 = ap0[1];
        float4 a10 = ap1[0], a11 = ap1[1];
        float af[16] = {a00.x,a00.y,a00.z,a00.w, a01.x,a01.y,a01.z,a01.w,
                        a10.x,a10.y,a10.z,a10.w, a11.x,a11.y,a11.z,a11.w};
        v16bf ahi, alo;
        #pragma unroll
        for (int i = 0; i < 16; ++i) {
            bf16_t h = f2bf(af[i]);
            ahi[i] = h;
            alo[i] = f2bf(af[i] - bf2f(h));
        }
        // ---- B: pre-swizzled contiguous fragments ----
        size_t fb = ((size_t)(kk >> 5) * ntiles + nt0) * 512 + (size_t)lane * 16;
        v16bf bh0 = *reinterpret_cast<const v16bf*>(Bhi + fb);
        v16bf bl0 = *reinterpret_cast<const v16bf*>(Blo + fb);
        v16bf bh1 = *reinterpret_cast<const v16bf*>(Bhi + fb + 512);
        v16bf bl1 = *reinterpret_cast<const v16bf*>(Blo + fb + 512);

        acc0 = __builtin_amdgcn_wmma_f32_16x16x32_bf16(false, ahi, false, bh0, (short)0, acc0, false, false);
        acc0 = __builtin_amdgcn_wmma_f32_16x16x32_bf16(false, alo, false, bh0, (short)0, acc0, false, false);
        acc0 = __builtin_amdgcn_wmma_f32_16x16x32_bf16(false, ahi, false, bl0, (short)0, acc0, false, false);
        acc1 = __builtin_amdgcn_wmma_f32_16x16x32_bf16(false, ahi, false, bh1, (short)0, acc1, false, false);
        acc1 = __builtin_amdgcn_wmma_f32_16x16x32_bf16(false, alo, false, bh1, (short)0, acc1, false, false);
        acc1 = __builtin_amdgcn_wmma_f32_16x16x32_bf16(false, ahi, false, bl1, (short)0, acc1, false, false);
    }

    // C/D layout: VGPR r -> row m0+r (lanes 0-15) / m0+8+r (lanes 16-31)
    const int rowoff = (lane >> 4) * 8;
    #pragma unroll
    for (int t = 0; t < 2; ++t) {
        const int ccol = (nt0 + t) * 16 + (lane & 15);
        const float bb = bias ? bias[ccol] : 0.0f;
        const v8f& acc = t ? acc1 : acc0;
        #pragma unroll
        for (int r = 0; r < 8; ++r) {
            int row = m0 + rowoff + r;
            if (row < M) {
                float v = acc[r] + bb;
                if (fuse_relu) v = fmaxf(v, 0.0f);
                C[(size_t)row * N + ccol] = v;
            }
        }
    }
}

// ---------------- elementwise / graph kernels ----------------
__global__ void k_fill_zero(float* __restrict__ p, long n) {
    long i = (long)blockIdx.x * blockDim.x + threadIdx.x;
    if (i < n) p[i] = 0.0f;
}

__global__ void k_deg_count(const int* __restrict__ dst, float* __restrict__ deg, int E) {
    int e = blockIdx.x * blockDim.x + threadIdx.x;
    if (e < E) atomicAdd(&deg[dst[e]], 1.0f);
}

__global__ void k_rsqrt_deg(float* __restrict__ dinv, int N) {
    int i = blockIdx.x * blockDim.x + threadIdx.x;
    if (i < N) dinv[i] = rsqrtf(dinv[i] + 1.0f);   // +1 self loop
}

// agg[i,c] = hw[i,c] * dinv[i]^2 + b[c]   (self-loop term + bias)
__global__ __launch_bounds__(256) void k_agg_init(
    const float* __restrict__ hw, const float* __restrict__ bias,
    const float* __restrict__ dinv, float* __restrict__ agg, int N)
{
    long t = (long)blockIdx.x * 256 + threadIdx.x;
    if (t < (long)N * 256) {
        int i = (int)(t >> 8), c = (int)(t & 255);
        float d = dinv[i];
        agg[t] = hw[t] * d * d + bias[c];
    }
}

// 64 threads per edge, 4 channels each: agg[dst] += hw[src] * dinv[src]*dinv[dst]
__global__ __launch_bounds__(256) void k_edge_agg(
    const float* __restrict__ hw, const int* __restrict__ src, const int* __restrict__ dst,
    const float* __restrict__ dinv, float* __restrict__ agg, int E)
{
    long t = (long)blockIdx.x * 256 + threadIdx.x;
    int e  = (int)(t >> 6);
    if (e >= E) return;
    int c0 = (int)(t & 63) << 2;
    int s = src[e], d = dst[e];
    float coef = dinv[s] * dinv[d];
    float4 hv = *reinterpret_cast<const float4*>(hw + (size_t)s * 256 + c0);
    float* ap = agg + (size_t)d * 256 + c0;
    atomicAdd(ap + 0, hv.x * coef);
    atomicAdd(ap + 1, hv.y * coef);
    atomicAdd(ap + 2, hv.z * coef);
    atomicAdd(ap + 3, hv.w * coef);
}

// out = relu(agg) + (res ? res : 0)
__global__ void k_finalize(const float* __restrict__ agg, const float* __restrict__ res,
                           float* __restrict__ out, long n)
{
    long i = (long)blockIdx.x * blockDim.x + threadIdx.x;
    if (i < n) out[i] = fmaxf(agg[i], 0.0f) + (res ? res[i] : 0.0f);
}

// fused layer-3 finalize + mean-pool scatter: sums[batch[i]] += relu(agg)+res
__global__ __launch_bounds__(256) void k_pool_scatter_fused(
    const float* __restrict__ agg, const float* __restrict__ res,
    const int* __restrict__ batch, float* __restrict__ sums, int N)
{
    long t = (long)blockIdx.x * 256 + threadIdx.x;
    if (t < (long)N * 256) {
        int i = (int)(t >> 8), c = (int)(t & 255);
        float v = fmaxf(agg[t], 0.0f) + res[t];
        atomicAdd(&sums[(size_t)batch[i] * 256 + c], v);
    }
}

__global__ void k_pool_cnt(const int* __restrict__ batch, float* __restrict__ cnt, int N) {
    int i = blockIdx.x * blockDim.x + threadIdx.x;
    if (i < N) atomicAdd(&cnt[batch[i]], 1.0f);
}

__global__ void k_pool_mean(const float* __restrict__ sums, const float* __restrict__ cnt,
                            float* __restrict__ g, long n)
{
    long i = (long)blockIdx.x * blockDim.x + threadIdx.x;
    if (i < n) g[i] = sums[i] / fmaxf(cnt[i >> 8], 1.0f);
}

// in-place LayerNorm over D=768 (3 values/thread, 256 threads/row)
__global__ __launch_bounds__(256) void k_layernorm(
    float* __restrict__ g, const float* __restrict__ gamma, const float* __restrict__ beta, int D)
{
    __shared__ float s1[256], s2[256];
    float* p = g + (size_t)blockIdx.x * D;
    float v[3]; float sum = 0.0f, sq = 0.0f;
    #pragma unroll
    for (int j = 0; j < 3; ++j) {
        int c = threadIdx.x + j * 256;
        v[j] = p[c]; sum += v[j]; sq += v[j] * v[j];
    }
    s1[threadIdx.x] = sum; s2[threadIdx.x] = sq;
    __syncthreads();
    for (int s = 128; s > 0; s >>= 1) {
        if (threadIdx.x < s) { s1[threadIdx.x] += s1[threadIdx.x + s]; s2[threadIdx.x] += s2[threadIdx.x + s]; }
        __syncthreads();
    }
    float mu  = s1[0] / (float)D;
    float var = s2[0] / (float)D - mu * mu;
    float rs  = rsqrtf(var + 1e-5f);
    #pragma unroll
    for (int j = 0; j < 3; ++j) {
        int c = threadIdx.x + j * 256;
        p[c] = (v[j] - mu) * rs * gamma[c] + beta[c];
    }
}

// ---------------- launcher ----------------
extern "C" void kernel_launch(void* const* d_in, const int* in_sizes, int n_in,
                              void* d_out, int out_size, void* d_ws, size_t ws_size,
                              hipStream_t stream)
{
    const float* x    = (const float*)d_in[0];
    const int*   eidx = (const int*)  d_in[1];
    const int*   batch= (const int*)  d_in[2];
    const float* W1   = (const float*)d_in[3];
    const float* b1   = (const float*)d_in[4];
    const float* W2   = (const float*)d_in[5];
    const float* b2   = (const float*)d_in[6];
    const float* W3   = (const float*)d_in[7];
    const float* b3   = (const float*)d_in[8];
    const float* P1   = (const float*)d_in[9];
    const float* pb1  = (const float*)d_in[10];
    const float* P2   = (const float*)d_in[11];
    const float* pb2  = (const float*)d_in[12];
    const float* ln_g = (const float*)d_in[13];
    const float* ln_b = (const float*)d_in[14];

    const int F_IN = 128, H = 256, D = 768, G = 1024;
    const int N = in_sizes[0] / F_IN;     // 50000 (multiple of 16)
    const int E = in_sizes[1] / 2;        // 800000
    const int* src = eidx;
    const int* dst = eidx + E;

    float* ws = (float*)d_ws;
    size_t off = 0;
    auto carve = [&](size_t nf) { float* p = ws + off; off += (nf + 63) & ~(size_t)63; return p; };
    auto carve_bf = [&](size_t nbf) { return (bf16_t*)carve((nbf + 1) / 2); };

    float* dinv = carve((size_t)N);
    float* hw   = carve((size_t)N * H);
    float* agg  = carve((size_t)N * H);
    float* h1   = carve((size_t)N * H);
    float* h2   = carve((size_t)N * H);
    float* sums = carve((size_t)G * H);
    float* cnt  = carve((size_t)G);
    float* gbuf = carve((size_t)G * H);
    float* g1   = carve((size_t)G * H);
    // pre-swizzled bf16 hi/lo weight planes
    bf16_t* W1h = carve_bf((size_t)F_IN * H); bf16_t* W1l = carve_bf((size_t)F_IN * H);
    bf16_t* W2h = carve_bf((size_t)H * H);    bf16_t* W2l = carve_bf((size_t)H * H);
    bf16_t* W3h = carve_bf((size_t)H * H);    bf16_t* W3l = carve_bf((size_t)H * H);
    bf16_t* P1h = carve_bf((size_t)H * H);    bf16_t* P1l = carve_bf((size_t)H * H);
    bf16_t* P2h = carve_bf((size_t)H * D);    bf16_t* P2l = carve_bf((size_t)H * D);
    float* out  = (float*)d_out;
    (void)ws_size; (void)n_in; (void)out_size;

    const long NH = (long)N * H;
    dim3 blk(256);

    // ---- weight pre-swizzle (tiny; independent of everything else) ----
    k_prep_w<<<(F_IN * H + 255) / 256, blk, 0, stream>>>(W1, W1h, W1l, F_IN, H);
    k_prep_w<<<(H * H + 255) / 256, blk, 0, stream>>>(W2, W2h, W2l, H, H);
    k_prep_w<<<(H * H + 255) / 256, blk, 0, stream>>>(W3, W3h, W3l, H, H);
    k_prep_w<<<(H * H + 255) / 256, blk, 0, stream>>>(P1, P1h, P1l, H, H);
    k_prep_w<<<(H * D + 255) / 256, blk, 0, stream>>>(P2, P2h, P2l, H, D);

    // ---- degrees -> dinv = rsqrt(deg+1) ----
    k_fill_zero<<<(N + 255) / 256, blk, 0, stream>>>(dinv, N);
    k_deg_count<<<(E + 255) / 256, blk, 0, stream>>>(dst, dinv, E);
    k_rsqrt_deg<<<(N + 255) / 256, blk, 0, stream>>>(dinv, N);

    dim3 gemmN(H / 256, (N + 15) / 16);          // node GEMMs: 256 cols/block
    int  aggBlocks  = (int)((NH + 255) / 256);
    int  edgeBlocks = (int)(((long)E * 64 + 255) / 256);

    // ---- layer 1: h1 = relu(gcn(x, W1, b1)) ----
    k_gemm_bf16x3<<<gemmN, blk, 0, stream>>>(x, W1h, W1l, nullptr, hw, N, F_IN, H, 0);
    k_agg_init<<<aggBlocks, blk, 0, stream>>>(hw, b1, dinv, agg, N);
    k_edge_agg<<<edgeBlocks, blk, 0, stream>>>(hw, src, dst, dinv, agg, E);
    k_finalize<<<aggBlocks, blk, 0, stream>>>(agg, nullptr, h1, NH);

    // ---- layer 2: h2 = relu(gcn(h1, W2, b2)) + h1 ----
    k_gemm_bf16x3<<<gemmN, blk, 0, stream>>>(h1, W2h, W2l, nullptr, hw, N, H, H, 0);
    k_agg_init<<<aggBlocks, blk, 0, stream>>>(hw, b2, dinv, agg, N);
    k_edge_agg<<<edgeBlocks, blk, 0, stream>>>(hw, src, dst, dinv, agg, E);
    k_finalize<<<aggBlocks, blk, 0, stream>>>(agg, h1, h2, NH);

    // ---- layer 3 (fused with mean-pool scatter): h3 = relu(gcn(h2,..)) + h2 ----
    k_gemm_bf16x3<<<gemmN, blk, 0, stream>>>(h2, W3h, W3l, nullptr, hw, N, H, H, 0);
    k_agg_init<<<aggBlocks, blk, 0, stream>>>(hw, b3, dinv, agg, N);
    k_edge_agg<<<edgeBlocks, blk, 0, stream>>>(hw, src, dst, dinv, agg, E);
    k_fill_zero<<<((long)G * H + 255) / 256, blk, 0, stream>>>(sums, (long)G * H);
    k_fill_zero<<<(G + 255) / 256, blk, 0, stream>>>(cnt, G);
    k_pool_scatter_fused<<<aggBlocks, blk, 0, stream>>>(agg, h2, batch, sums, N);
    k_pool_cnt<<<(N + 255) / 256, blk, 0, stream>>>(batch, cnt, N);
    k_pool_mean<<<((long)G * H + 255) / 256, blk, 0, stream>>>(sums, cnt, gbuf, (long)G * H);

    // ---- projection MLP + LayerNorm ----
    dim3 gemmP1(H / 256, G / 16);                // 1024x256 @ 256x256, relu+bias
    k_gemm_bf16x3<<<gemmP1, blk, 0, stream>>>(gbuf, P1h, P1l, pb1, g1, G, H, H, 1);
    dim3 gemmP2(D / 256, G / 16);                // 1024x256 @ 256x768 -> d_out
    k_gemm_bf16x3<<<gemmP2, blk, 0, stream>>>(g1, P2h, P2l, pb2, out, G, H, D, 0);
    k_layernorm<<<G, blk, 0, stream>>>(out, ln_g, ln_b, D);
}
